// SetConvGridDecoder_21105469292681
// MI455X (gfx1250) — matrix operand order repaired and checked by reading, not compiled, not run on hardware
//
#include <hip/hip_runtime.h>
#include <hip/hip_bf16.h>

typedef __attribute__((ext_vector_type(16))) _Float16 v16h;
typedef __attribute__((ext_vector_type(8)))  float    v8f;
typedef __attribute__((ext_vector_type(2)))  float    v2f;

#define M_BATCH 8
#define NC_MAX  4096   // gh*gw from the reference setup

#if defined(__has_builtin)
#if __has_builtin(__builtin_amdgcn_wmma_f32_16x16x4_f32)
#define HAVE_WMMA_F32X4 1
#endif
#if __has_builtin(__builtin_amdgcn_exp2f)
#define EXP2(x) __builtin_amdgcn_exp2f(x)   // raw v_exp_f32; denorm flush is
#else                                       // fine (weights ~0 there anyway)
#define EXP2(x) exp2f(x)
#endif
#endif
#ifndef EXP2
#define EXP2(x) exp2f(x)
#endif

// ---------------------------------------------------------------------------
// Pre-pass: zc [m, nc, dz] f32  ->  zcT [m, dz, nc] f16
// K-contiguous f16 B operand: each lane's 16-half fragment = two b128 loads.
// ---------------------------------------------------------------------------
__global__ void zc_to_f16_T(const float* __restrict__ zc,
                            _Float16* __restrict__ zcT,
                            int nc, int dz, long total) {
  long i = (long)blockIdx.x * blockDim.x + threadIdx.x;
  if (i >= total) return;
  int  c = (int)(i % nc);
  long t = i / nc;
  int  n = (int)(t % dz);
  long b = t / dz;
  zcT[i] = (_Float16)zc[((long)b * nc + c) * (long)dz + n];
}

// ---------------------------------------------------------------------------
// Fused RBF + GEMM + residual. Block = 256 thr = 8 waves; wave: 16 rows x 64.
//
// Distance tiles via V_WMMA_F32_16X16X4_F32 with the norm biases folded into
// the two spare K slots (dx=2, K=4):
//   K0,K1: <ac~, at~>      (coords scaled by sqrt(2g), g = 0.5*log2e/ls^2)
//   K2   : 1 * (-|at~|^2/2)
//   K3   : (-|ac~|^2/2) * 1
// => D = -|at~ - ac~|^2 / 2  (log2-exponent, <= 0: no overflow), C = 0.
// Layout gift: K2/K3 of both A and B live in lanes 16-31, and the f32 D tile
// (lane->target, v->context 8*half+v) is exactly the f16 A-fragment layout,
// so D feeds V_WMMA_F32_16X16X32_F16 with no cross-lane movement.
// ---------------------------------------------------------------------------
__global__ __launch_bounds__(256, 1)
void setconv_fused(const float* __restrict__ xc,       // [m, nc, 2]
                   const _Float16* __restrict__ zcT,   // [m, dz, nc] f16
                   const float* __restrict__ xt,       // [m, nt, 2]
                   const float* __restrict__ zt,       // [m, nt, dz]
                   const float* __restrict__ lp,       // [2]
                   float* __restrict__ out,            // [m, nt, dz]
                   int nc, int nt, int dz) {
  // Per-lane-half staged A operand of the distance WMMA (64 KB):
  //   s_ab[0*NC + i] = (r0*xc0, r1*xc1)      -> lanes 0-15  (K0,K1)
  //   s_ab[1*NC + i] = (1.0, -|ac~|^2/2)     -> lanes 16-31 (K2,K3)
  __shared__ float2 s_ab[2 * NC_MAX];

  const int b    = blockIdx.y;
  const int tid  = threadIdx.x;
  const int lane = tid & 31;
  const int wave = tid >> 5;
  const int half = lane >> 4;
  const int ln15 = lane & 15;

  // ls = 1e-5 + softplus(lp);  r_d = sqrt(2*g_d) = sqrt(log2 e)/ls_d
  float r0, r1;
  {
    float ls0 = 1e-5f + log1pf(__expf(lp[0]));
    float ls1 = 1e-5f + log1pf(__expf(lp[1]));
    const float sl2e = 1.2011224087864498f;  // sqrt(log2(e))
    r0 = sl2e / ls0;
    r1 = sl2e / ls1;
  }

  {
    const float2* xc2 = (const float2*)xc + (long)b * nc;
    for (int i = tid; i < nc; i += 256) {
      float2 p = xc2[i];
      float2 q = make_float2(r0 * p.x, r1 * p.y);
      s_ab[i]          = q;
      s_ab[NC_MAX + i] = make_float2(1.0f, -0.5f * (q.x * q.x + q.y * q.y));
    }
  }
  __syncthreads();

  const int row_base = blockIdx.x * 128 + wave * 16;
  const float2 xtp = ((const float2*)xt)[(long)b * nt + row_base + ln15];
  const float at0 = r0 * xtp.x, at1 = r1 * xtp.y;
  const float ntn = -0.5f * (at0 * at0 + at1 * at1);   // -|at~|^2/2

#if HAVE_WMMA_F32X4
  // B fragment (4x16 f32), loop-invariant:
  // lanes 0-15: (K0,K1)=(at0,at1); lanes 16-31: (K2,K3)=(ntn, 1).
  v2f bdist;
  bdist[0] = half ? ntn : at0;
  bdist[1] = half ? 1.0f : at1;
#endif

  v8f acc0 = {}, acc1 = {}, acc2 = {}, acc3 = {};

  const _Float16* bcol = zcT + ((long)b * dz + ln15) * nc;
  const long bstride = (long)16 * nc;
  const float2* s_lane = s_ab + half * NC_MAX + ln15;  // per-half A source

  for (int kc = 0; kc < nc; kc += 32) {
    v16h a;
#if HAVE_WMMA_F32X4
    // One ds_load_2addr_b64: contexts kc+ln15 and kc+16+ln15.
    float2 pa = s_lane[kc];
    float2 pb = s_lane[kc + 16];
    v2f ad1, ad2;
    ad1[0] = pa.x;  ad1[1] = pa.y;
    ad2[0] = pb.x;  ad2[1] = pb.y;

    v8f zc8 = {};
    v8f d1 = __builtin_amdgcn_wmma_f32_16x16x4_f32(false, ad1, false, bdist,
                                                   (short)0, zc8, false, false);
    v8f d2 = __builtin_amdgcn_wmma_f32_16x16x4_f32(false, ad2, false, bdist,
                                                   (short)0, zc8, false, false);
#pragma unroll
    for (int v = 0; v < 8; ++v) {
      a[v]     = (_Float16)EXP2(d1[v]);  // ctx kc    + 8*half + v == elem v
      a[8 + v] = (_Float16)EXP2(d2[v]);  // ctx kc+16 + 8*half + v == elem 8+v
    }
#else
    // Fallback: direct per-point evaluation on prescaled coords.
    const int k0 = kc + half * 8;
#pragma unroll
    for (int i = 0; i < 8; ++i) {
      float2 p = s_ab[k0 + i];
      float u = at0 - p.x, w = at1 - p.y;
      a[i] = (_Float16)EXP2(-0.5f * (u * u + w * w));
      float2 q = s_ab[k0 + 16 + i];
      float u2 = at0 - q.x, w2 = at1 - q.y;
      a[8 + i] = (_Float16)EXP2(-0.5f * (u2 * u2 + w2 * w2));
    }
#endif

    typedef __attribute__((ext_vector_type(8))) _Float16 v8h_t;
    const long koff = kc + half * 8;
    const _Float16* p0 = bcol + 0 * bstride + koff;
    const _Float16* p1 = bcol + 1 * bstride + koff;
    const _Float16* p2 = bcol + 2 * bstride + koff;
    const _Float16* p3 = bcol + 3 * bstride + koff;
    v16h b0 = __builtin_shufflevector(*(const v8h_t*)p0, *(const v8h_t*)(p0 + 16), 0,1,2,3,4,5,6,7,8,9,10,11,12,13,14,15);
    v16h b1 = __builtin_shufflevector(*(const v8h_t*)p1, *(const v8h_t*)(p1 + 16), 0,1,2,3,4,5,6,7,8,9,10,11,12,13,14,15);
    v16h b2 = __builtin_shufflevector(*(const v8h_t*)p2, *(const v8h_t*)(p2 + 16), 0,1,2,3,4,5,6,7,8,9,10,11,12,13,14,15);
    v16h b3 = __builtin_shufflevector(*(const v8h_t*)p3, *(const v8h_t*)(p3 + 16), 0,1,2,3,4,5,6,7,8,9,10,11,12,13,14,15);

    acc0 = __builtin_amdgcn_wmma_f32_16x16x32_f16(false, a, false, b0,
                                                  (short)0, acc0, false, false);
    acc1 = __builtin_amdgcn_wmma_f32_16x16x32_f16(false, a, false, b1,
                                                  (short)0, acc1, false, false);
    acc2 = __builtin_amdgcn_wmma_f32_16x16x32_f16(false, a, false, b2,
                                                  (short)0, acc2, false, false);
    acc3 = __builtin_amdgcn_wmma_f32_16x16x32_f16(false, a, false, b3,
                                                  (short)0, acc3, false, false);
  }

  // Epilogue: out = zt + acc. C/D layout: lane -> N, vgpr v -> M = v + 8*half.
  const long orow = (long)b * nt + row_base + half * 8;
#pragma unroll
  for (int v = 0; v < 8; ++v) {
    long base = (orow + v) * (long)dz + ln15;
    out[base +  0] = zt[base +  0] + acc0[v];
    out[base + 16] = zt[base + 16] + acc1[v];
    out[base + 32] = zt[base + 32] + acc2[v];
    out[base + 48] = zt[base + 48] + acc3[v];
  }
}

extern "C" void kernel_launch(void* const* d_in, const int* in_sizes, int n_in,
                              void* d_out, int out_size, void* d_ws, size_t ws_size,
                              hipStream_t stream) {
  const float* xc = (const float*)d_in[0];
  const float* zc = (const float*)d_in[1];
  const float* xt = (const float*)d_in[2];
  const float* zt = (const float*)d_in[3];
  const float* lp = (const float*)d_in[4];
  float* out = (float*)d_out;

  const int m  = M_BATCH;                 // fixed by the reference setup
  const int dx = in_sizes[4];             // 2
  const int nc = in_sizes[0] / (m * dx);  // 4096
  const int nt = in_sizes[2] / (m * dx);  // 2048
  const int dz = in_sizes[1] / (m * nc);  // 64

  _Float16* zcT = (_Float16*)d_ws;        // 4 MB workspace
  const long total = (long)m * dz * nc;
  zc_to_f16_T<<<(unsigned)((total + 255) / 256), 256, 0, stream>>>(zc, zcT, nc,
                                                                   dz, total);

  dim3 grid((unsigned)(nt / 128), (unsigned)m);
  setconv_fused<<<grid, 256, 0, stream>>>(xc, zcT, xt, zt, lp, out, nc, nt, dz);
}
